// pointconv_52725018526360
// MI455X (gfx1250) — matrix operand order repaired and checked by reading, not compiled, not run on hardware
//
#include <hip/hip_runtime.h>

#define BB 4
#define NN 4096

typedef __attribute__((ext_vector_type(16))) _Float16 v16h;
typedef __attribute__((ext_vector_type(8)))  _Float16 v8h;
typedef __attribute__((ext_vector_type(8)))  float    v8f;

// ======================================================================
// Device kernels
// ======================================================================

// (B,6,N) -> pts (B,N,6) and l0_xyz (B,N,3)
__global__ void k_make_pts(const float* __restrict__ xyz,
                           float* __restrict__ pts, float* __restrict__ l0) {
  int t = blockIdx.x * blockDim.x + threadIdx.x;
  if (t >= BB * NN) return;
  int b = t / NN, n = t % NN;
#pragma unroll
  for (int c = 0; c < 6; ++c) {
    float v = xyz[((size_t)b * 6 + c) * NN + n];
    pts[((size_t)b * NN + n) * 6 + c] = v;
    if (c < 3) l0[((size_t)b * NN + n) * 3 + c] = v;
  }
}

// inverse kernel-density per point
__global__ void k_density(const float* __restrict__ xyz, int Np, float bw,
                          float* __restrict__ invd) {
  __shared__ float tile[256 * 3];
  int b = blockIdx.y;
  int i = blockIdx.x * 256 + threadIdx.x;
  const float* X = xyz + (size_t)b * Np * 3;
  bool act = (i < Np);
  float px = 0.f, py = 0.f, pz = 0.f;
  if (act) { px = X[i * 3]; py = X[i * 3 + 1]; pz = X[i * 3 + 2]; }
  float inv2 = 1.f / (2.f * bw * bw);
  float acc = 0.f;
  for (int j0 = 0; j0 < Np; j0 += 256) {
    int j = j0 + threadIdx.x;
    if (j < Np) {
      tile[threadIdx.x * 3 + 0] = X[j * 3 + 0];
      tile[threadIdx.x * 3 + 1] = X[j * 3 + 1];
      tile[threadIdx.x * 3 + 2] = X[j * 3 + 2];
    }
    __syncthreads();
    int cnt = min(256, Np - j0);
    for (int jj = 0; jj < cnt; ++jj) {
      float dx = tile[jj * 3] - px, dy = tile[jj * 3 + 1] - py, dz = tile[jj * 3 + 2] - pz;
      acc += __expf(-(dx * dx + dy * dy + dz * dz) * inv2);
    }
    __syncthreads();
  }
  if (act) {
    float density = acc / ((float)Np * 2.5f * bw);
    invd[(size_t)b * Np + i] = 1.f / density;
  }
}

// Farthest point sampling: one workgroup per batch, LDS distance array.
__global__ void k_fps(const float* __restrict__ xyz, int Np, int npoint,
                      int* __restrict__ fidx, float* __restrict__ new_xyz) {
  __shared__ float dist[NN];
  __shared__ float s_redv[8];
  __shared__ int   s_redi[8];
  __shared__ int   s_far;
  __shared__ float s_cen[3];
  int b = blockIdx.x;
  const float* xb = xyz + (size_t)b * Np * 3;
  int tid = threadIdx.x;
  for (int i = tid; i < Np; i += blockDim.x) dist[i] = 1e10f;
  if (tid == 0) s_far = 0;
  __syncthreads();
  for (int t = 0; t < npoint; ++t) {
    int far = s_far;
    if (tid == 0) {
      fidx[(size_t)b * npoint + t] = far;
      s_cen[0] = xb[far * 3]; s_cen[1] = xb[far * 3 + 1]; s_cen[2] = xb[far * 3 + 2];
      new_xyz[((size_t)b * npoint + t) * 3 + 0] = s_cen[0];
      new_xyz[((size_t)b * npoint + t) * 3 + 1] = s_cen[1];
      new_xyz[((size_t)b * npoint + t) * 3 + 2] = s_cen[2];
    }
    __syncthreads();
    float cx = s_cen[0], cy = s_cen[1], cz = s_cen[2];
    float bestd = -1.f; int besti = 0;
    for (int i = tid; i < Np; i += blockDim.x) {
      float dx = xb[i * 3] - cx, dy = xb[i * 3 + 1] - cy, dz = xb[i * 3 + 2] - cz;
      float d = dx * dx + dy * dy + dz * dz;
      float nd = fminf(dist[i], d);
      dist[i] = nd;
      if (nd > bestd || (nd == bestd && i < besti)) { bestd = nd; besti = i; }
    }
    for (int o = 16; o > 0; o >>= 1) {
      float od = __shfl_down(bestd, o);
      int   oi = __shfl_down(besti, o);
      if (od > bestd || (od == bestd && oi < besti)) { bestd = od; besti = oi; }
    }
    if ((tid & 31) == 0) { s_redv[tid >> 5] = bestd; s_redi[tid >> 5] = besti; }
    __syncthreads();
    if (tid == 0) {
      float bd = s_redv[0]; int bi = s_redi[0];
      int nw = blockDim.x >> 5;
      for (int w = 1; w < nw; ++w)
        if (s_redv[w] > bd || (s_redv[w] == bd && s_redi[w] < bi)) { bd = s_redv[w]; bi = s_redi[w]; }
      s_far = bi;
    }
    __syncthreads();
  }
}

// kNN (smallest-K squared distances, ascending; ties keep earlier index).
template <int K>
__global__ void k_knn(const float* __restrict__ q, int nq,
                      const float* __restrict__ src, int ns,
                      int* __restrict__ oidx, float* __restrict__ od) {
  __shared__ float sd[64][K];
  __shared__ int   si[64][K];
  int b = blockIdx.y, t = threadIdx.x;
  int qi = blockIdx.x * 64 + t;
  bool act = (qi < nq);
  float qx = 0.f, qy = 0.f, qz = 0.f;
  if (act) {
    const float* Q = q + ((size_t)b * nq + qi) * 3;
    qx = Q[0]; qy = Q[1]; qz = Q[2];
  }
  for (int k = 0; k < K; ++k) { sd[t][k] = 3.4e38f; si[t][k] = 0; }
  const float* S = src + (size_t)b * ns * 3;
  for (int j = 0; j < ns; ++j) {
    float dx = S[j * 3] - qx, dy = S[j * 3 + 1] - qy, dz = S[j * 3 + 2] - qz;
    float d = dx * dx + dy * dy + dz * dz;
    if (act && d < sd[t][K - 1]) {
      int pos = K - 1;
      while (pos > 0 && sd[t][pos - 1] > d) {
        sd[t][pos] = sd[t][pos - 1]; si[t][pos] = si[t][pos - 1]; --pos;
      }
      sd[t][pos] = d; si[t][pos] = j;
    }
  }
  if (act) {
    for (int k = 0; k < K; ++k) {
      oidx[((size_t)b * nq + qi) * K + k] = si[t][k];
      od  [((size_t)b * nq + qi) * K + k] = sd[t][k];
    }
  }
}

// zero-fill an f16 buffer
__global__ void k_zero16(_Float16* __restrict__ p, size_t n) {
  size_t t = (size_t)blockIdx.x * blockDim.x + threadIdx.x;
  if (t < n) p[t] = (_Float16)0.f;
}

// Pack weights: W (K,N) f32 -> Wt (Np16, Kp) f16 transposed, zero padded.
__global__ void k_pack_w(const float* __restrict__ W, _Float16* __restrict__ Wt,
                         int K, int N, int Kp, int Np16) {
  int t = blockIdx.x * blockDim.x + threadIdx.x;
  int tot = Np16 * Kp;
  if (t >= tot) return;
  int n = t / Kp, k = t % Kp;
  float v = (k < K && n < N) ? W[(size_t)k * N + n] : 0.f;
  Wt[t] = (_Float16)v;
}

// build feats0 = [g_norm | gather(points)] as packed f16 (tot x Kp),
// plus packed f16 g_norm (tot x 32, cols 3..31 = 0)
__global__ void k_gather_feats(const float* __restrict__ xyz,
                               const float* __restrict__ new_xyz,
                               const float* __restrict__ points,
                               const int* __restrict__ knn,
                               int np, int NpSrc, int Cin, int Kp,
                               _Float16* __restrict__ featspk,
                               _Float16* __restrict__ gnormpk) {
  int idx = blockIdx.x * blockDim.x + threadIdx.x;
  int tot = BB * np * 32;
  if (idx >= tot) return;
  int s = idx & 31; (void)s;
  int bp = idx >> 5;
  int p = bp % np, b = bp / np;
  int j = knn[(size_t)bp * 32 + (idx & 31)];
  float gx = xyz[((size_t)b * NpSrc + j) * 3 + 0] - new_xyz[((size_t)b * np + p) * 3 + 0];
  float gy = xyz[((size_t)b * NpSrc + j) * 3 + 1] - new_xyz[((size_t)b * np + p) * 3 + 1];
  float gz = xyz[((size_t)b * NpSrc + j) * 3 + 2] - new_xyz[((size_t)b * np + p) * 3 + 2];
  _Float16* G = gnormpk + (size_t)idx * 32;
  G[0] = (_Float16)gx; G[1] = (_Float16)gy; G[2] = (_Float16)gz;
  for (int c = 3; c < 32; ++c) G[c] = (_Float16)0.f;
  _Float16* F = featspk + (size_t)idx * Kp;
  F[0] = (_Float16)gx; F[1] = (_Float16)gy; F[2] = (_Float16)gz;
  const float* P = points + ((size_t)b * NpSrc + j) * Cin;
  for (int c = 0; c < Cin; ++c) F[3 + c] = (_Float16)P[c];
  for (int c = 3 + Cin; c < Kp; ++c) F[c] = (_Float16)0.f;
}

// dscale input: gathered inv-density / group max, packed f16 (tot x 32, col0)
__global__ void k_dscale_in(const float* __restrict__ invd, const int* __restrict__ knn,
                            int np, int NpSrc, _Float16* __restrict__ dsin) {
  int t = blockIdx.x * blockDim.x + threadIdx.x;
  int tot = BB * np * 32;
  if (t >= tot) return;
  int s = t & 31;
  int bp = t >> 5;
  int b = bp / np;
  const int* I = knn + (size_t)bp * 32;
  float mx = 0.f, v = 0.f;
#pragma unroll
  for (int j = 0; j < 32; ++j) {
    float u = invd[(size_t)b * NpSrc + I[j]];
    if (j == s) v = u;
    mx = fmaxf(mx, u);
  }
  _Float16* o = dsin + (size_t)t * 32;
  o[0] = (_Float16)(v / mx);
  for (int c = 1; c < 32; ++c) o[c] = (_Float16)0.f;
}

// feats[row, c] *= ds[row]   (f32)
__global__ void k_scale(float* __restrict__ feats, const float* __restrict__ ds,
                        int C, int rows) {
  size_t e = (size_t)blockIdx.x * blockDim.x + threadIdx.x;
  size_t tot = (size_t)rows * C;
  if (e >= tot) return;
  feats[e] *= ds[e / C];
}

// ---------------------------------------------------------------------
// Fused GEMM (+bias +BN +act) via v_wmma_f32_16x16x32_f16.
// Apk: (M, Kp) f16 row-major, K padded to mult of 32 with zeros.
// Wt:  (Np16, Kp) f16, weights transposed + zero padded.
// Each wave computes a 64x16 tile (4 M-tiles); 4 waves/block = 4 N-tiles.
// All loads are 128-bit, no guards in the K loop (padding is zero).
// act: 0=relu, 1=sigmoid, 2=none
// ---------------------------------------------------------------------
__global__ void k_gemm(const _Float16* __restrict__ Apk, int Kp,
                       const _Float16* __restrict__ Wt,
                       const float* __restrict__ bias,
                       const float* __restrict__ bng, const float* __restrict__ bnbe,
                       const float* __restrict__ bnm, const float* __restrict__ bnv,
                       float* __restrict__ out32, _Float16* __restrict__ out16, int KpOut,
                       int M, int N, int act, int has_bn) {
  int lane = threadIdx.x & 31;
  int wave = threadIdx.x >> 5;
  int ntiles = (N + 15) >> 4;
  int tn = blockIdx.y * 4 + wave;
  if (tn >= ntiles) return;                 // wave-uniform exit
  int m0 = blockIdx.x * 64;                 // 4 consecutive M tiles
  int col = tn * 16 + (lane & 15);
  // ISA fragment layouts (16-bit):
  //   A: lanes<16 hold K {0-7,16-23}, lanes>=16 hold K {8-15,24-31}
  //   B: lanes<16 hold K {0-7,16-23}, lanes>=16 hold K {8-15,24-31} (column = lane&15)
  int khiA = (lane & 16) ? 8 : 0;
  int khiB = (lane & 16) ? 8 : 0;
  const v8h* Bp  = (const v8h*)(Wt + (size_t)col * Kp);
  int r0 = m0 + (lane & 15);
  const v8h* Ap0 = (const v8h*)(Apk + (size_t)(r0 +  0) * Kp);
  const v8h* Ap1 = (const v8h*)(Apk + (size_t)(r0 + 16) * Kp);
  const v8h* Ap2 = (const v8h*)(Apk + (size_t)(r0 + 32) * Kp);
  const v8h* Ap3 = (const v8h*)(Apk + (size_t)(r0 + 48) * Kp);
  v8f acc0 = {}, acc1 = {}, acc2 = {}, acc3 = {};
  for (int kk = 0; kk < Kp; kk += 32) {
    int ia = (kk + khiA) >> 3;              // v8h index: chunks at +0 and +16 halves
    int ib = (kk + khiB) >> 3;
    __builtin_prefetch(Wt + (size_t)col * Kp + kk + 64, 0, 3);
    v8h bl = Bp[ib], bh = Bp[ib + 2];
    v16h bf = __builtin_shufflevector(bl, bh, 0,1,2,3,4,5,6,7,8,9,10,11,12,13,14,15);
    v16h a0 = __builtin_shufflevector(Ap0[ia], Ap0[ia + 2], 0,1,2,3,4,5,6,7,8,9,10,11,12,13,14,15);
    v16h a1 = __builtin_shufflevector(Ap1[ia], Ap1[ia + 2], 0,1,2,3,4,5,6,7,8,9,10,11,12,13,14,15);
    v16h a2 = __builtin_shufflevector(Ap2[ia], Ap2[ia + 2], 0,1,2,3,4,5,6,7,8,9,10,11,12,13,14,15);
    v16h a3 = __builtin_shufflevector(Ap3[ia], Ap3[ia + 2], 0,1,2,3,4,5,6,7,8,9,10,11,12,13,14,15);
    acc0 = __builtin_amdgcn_wmma_f32_16x16x32_f16(false, a0, false, bf, (short)0, acc0, false, false);
    acc1 = __builtin_amdgcn_wmma_f32_16x16x32_f16(false, a1, false, bf, (short)0, acc1, false, false);
    acc2 = __builtin_amdgcn_wmma_f32_16x16x32_f16(false, a2, false, bf, (short)0, acc2, false, false);
    acc3 = __builtin_amdgcn_wmma_f32_16x16x32_f16(false, a3, false, bf, (short)0, acc3, false, false);
  }
  // epilogue: y = (acc + bias) [*bn_scale + bn_shift] [act]
  float bi = 0.f, sc = 1.f, sh = 0.f;
  bool colOK = col < N;
  if (colOK) {
    bi = bias[col];
    if (has_bn) {
      sc = bng[col] * rsqrtf(bnv[col] + 1e-5f);
      sh = bnbe[col] - bnm[col] * sc;
    }
  }
  int mofs = (lane & 16) ? 8 : 0;
  v8f accs[4] = {acc0, acc1, acc2, acc3};
#pragma unroll
  for (int tile = 0; tile < 4; ++tile) {
#pragma unroll
    for (int r = 0; r < 8; ++r) {
      int m = m0 + tile * 16 + r + mofs;
      float y = accs[tile][r] + bi;
      if (has_bn) y = y * sc + sh;
      if (act == 0)      y = fmaxf(y, 0.f);
      else if (act == 1) y = 1.f / (1.f + __expf(-y));
      if (out32 && colOK) out32[(size_t)m * N + col] = y;
      if (out16) out16[(size_t)m * KpOut + col] = colOK ? (_Float16)y : (_Float16)0.f;
    }
  }
}

// einsum('bpsc,bpsk->bpck'): agg[c,k] = sum_s feats[s,c]*w[s,k] per point.
// Output written as packed f16 (bp, C*16).
__global__ void k_einsum(const float* __restrict__ feats, const float* __restrict__ wnet,
                         _Float16* __restrict__ aggpk, int C) {
  int lane = threadIdx.x & 31;
  int wave = threadIdx.x >> 5;
  int bp = blockIdx.y;
  int ct = blockIdx.x * 4 + wave;
  if (ct * 16 >= C) return;                 // wave-uniform
  const float* F = feats + (size_t)bp * 32 * C;
  const float* Wn = wnet + (size_t)bp * 32 * 16;
  int cidx = ct * 16 + (lane & 15);
  int nidx = lane & 15;
  int khiA = (lane & 16) ? 8 : 0;
  int khiB = (lane & 16) ? 16 : 0;
  v16h a, b;
#pragma unroll
  for (int e = 0; e < 16; ++e) {
    int sa = khiA + (e < 8 ? e : e + 8);
    int sb = khiA + (e < 8 ? e : e + 8);
    a[e] = (_Float16)F[(size_t)sa * C + cidx];
    b[e] = (_Float16)Wn[sb * 16 + nidx];
  }
  v8f acc = {};
  acc = __builtin_amdgcn_wmma_f32_16x16x32_f16(false, a, false, b, (short)0, acc, false, false);
  _Float16* O = aggpk + (size_t)bp * C * 16;
#pragma unroll
  for (int r = 0; r < 8; ++r) {
    int c = ct * 16 + r + ((lane & 16) ? 8 : 0);
    O[(size_t)c * 16 + nidx] = (_Float16)acc[r];
  }
}

// 3-NN inverse-distance interpolation + channel concat -> packed f16 (tot x Kp)
__global__ void k_interp_concat(const float* __restrict__ p1, int C1,
                                const float* __restrict__ p2, int C2,
                                const int* __restrict__ idx3, const float* __restrict__ d3,
                                int n1, int n2, int Kp, _Float16* __restrict__ out) {
  int t = blockIdx.x * blockDim.x + threadIdx.x;
  int tot = BB * n1;
  if (t >= tot) return;
  int b = t / n1;
  const int* I = idx3 + (size_t)t * 3;
  const float* D = d3 + (size_t)t * 3;
  float w0 = 1.f / (D[0] + 1e-8f);
  float w1 = 1.f / (D[1] + 1e-8f);
  float w2 = 1.f / (D[2] + 1e-8f);
  float ws = w0 + w1 + w2;
  w0 /= ws; w1 /= ws; w2 /= ws;
  _Float16* o = out + (size_t)t * Kp;
  for (int c = 0; c < C1; ++c) o[c] = (_Float16)p1[(size_t)t * C1 + c];
  const float* q0 = p2 + ((size_t)b * n2 + I[0]) * C2;
  const float* q1 = p2 + ((size_t)b * n2 + I[1]) * C2;
  const float* q2 = p2 + ((size_t)b * n2 + I[2]) * C2;
  for (int c = 0; c < C2; ++c)
    o[C1 + c] = (_Float16)(w0 * q0[c] + w1 * q1[c] + w2 * q2[c]);
  for (int c = C1 + C2; c < Kp; ++c) o[c] = (_Float16)0.f;
}

// log_softmax over 2 classes + transpose (B,N,2) -> (B,2,N)
__global__ void k_logsoftmax_T(const float* __restrict__ logits, float* __restrict__ out) {
  int t = blockIdx.x * blockDim.x + threadIdx.x;
  if (t >= BB * NN) return;
  int b = t / NN, n = t % NN;
  float a = logits[(size_t)t * 2 + 0];
  float c = logits[(size_t)t * 2 + 1];
  float mx = fmaxf(a, c);
  float lse = mx + logf(__expf(a - mx) + __expf(c - mx));
  out[(size_t)b * 2 * NN + 0 * NN + n] = a - lse;
  out[(size_t)b * 2 * NN + 1 * NN + n] = c - lse;
}

// ======================================================================
// Host side
// ======================================================================

struct Lin { const float* W; const float* b; };
struct Bn  { const float* be; const float* g; const float* m; const float* v; };
struct Layer { Bn bn; Lin lin; };
struct SAP { Bn bn_linear; Layer dens[3]; Lin linear; Layer mlp[3]; Layer wnet[3]; };

struct WS {
  _Float16 *fpkA, *fpkB, *dspkA, *dspkB, *wpkA, *wpkB, *gnormpk, *aggpk, *wtpk, *p0pk, *hpk;
  float *feats32, *wnet32, *ds32, *invd, *knn_d, *knn3_d;
  int *fidx, *knn_idx, *knn3_idx;
};

static inline int ru32(int x) { return (x + 31) & ~31; }
static inline int ru16(int x) { return (x + 15) & ~15; }

// Pack weights + launch fused WMMA GEMM.  M must be a multiple of 64 (all are).
static void gemm(hipStream_t s, WS& w, const _Float16* Apk, int K, int Kp,
                 const Lin& lin, const Bn* bn, float* out32, _Float16* out16,
                 int M, int N, int act) {
  int Np16 = ru16(N);
  int wtot = Np16 * Kp;
  k_pack_w<<<(wtot + 255) / 256, 256, 0, s>>>(lin.W, w.wtpk, K, N, Kp, Np16);
  dim3 grid(M / 64, (Np16 / 16 + 3) / 4);
  k_gemm<<<grid, 128, 0, s>>>(Apk, Kp, w.wtpk, lin.b,
      bn ? bn->g : nullptr, bn ? bn->be : nullptr,
      bn ? bn->m : nullptr, bn ? bn->v : nullptr,
      out32, out16, ru32(N), M, N, act, bn ? 1 : 0);
}

static void run_sa(hipStream_t s, const SAP& P,
                   const float* xyz_in, int Np, const float* pts_in, int Cpts,
                   int npoint, float bw, const int* mlpd,
                   float* out_xyz, float* out_feat, WS& w) {
  k_density<<<dim3((Np + 255) / 256, BB), 256, 0, s>>>(xyz_in, Np, bw, w.invd);
  k_fps<<<BB, 256, 0, s>>>(xyz_in, Np, npoint, w.fidx, out_xyz);
  k_knn<32><<<dim3((npoint + 63) / 64, BB), 64, 0, s>>>(out_xyz, npoint, xyz_in, Np,
                                                        w.knn_idx, w.knn_d);
  int tot = BB * npoint * 32;
  int Cf = 3 + Cpts, Kp0 = ru32(Cf);
  k_gather_feats<<<(tot + 255) / 256, 256, 0, s>>>(xyz_in, out_xyz, pts_in, w.knn_idx,
                                                   npoint, Np, Cpts, Kp0, w.fpkA, w.gnormpk);
  // feature MLP chain: fpkA -> fpkB -> fpkA -> feats32 (f32)
  gemm(s, w, w.fpkA, Cf, Kp0, P.mlp[0].lin, &P.mlp[0].bn, nullptr, w.fpkB, tot, mlpd[0], 0);
  gemm(s, w, w.fpkB, mlpd[0], ru32(mlpd[0]), P.mlp[1].lin, &P.mlp[1].bn, nullptr, w.fpkA, tot, mlpd[1], 0);
  gemm(s, w, w.fpkA, mlpd[1], ru32(mlpd[1]), P.mlp[2].lin, &P.mlp[2].bn, w.feats32, nullptr, tot, mlpd[2], 0);
  // density-scale net: 1 -> 16 -> 8 -> 1(sigmoid)
  k_dscale_in<<<(tot + 255) / 256, 256, 0, s>>>(w.invd, w.knn_idx, npoint, Np, w.dspkA);
  k_zero16<<<(unsigned)(((size_t)tot * 32 + 255) / 256), 256, 0, s>>>(w.dspkB, (size_t)tot * 32);
  gemm(s, w, w.dspkA, 1, 32, P.dens[0].lin, &P.dens[0].bn, nullptr, w.dspkB, tot, 16, 0);
  gemm(s, w, w.dspkB, 16, 32, P.dens[1].lin, &P.dens[1].bn, nullptr, w.dspkA, tot, 8, 0);
  gemm(s, w, w.dspkA, 8, 32, P.dens[2].lin, &P.dens[2].bn, w.ds32, nullptr, tot, 1, 1);
  int C = mlpd[2];
  {
    size_t n = (size_t)tot * C;
    k_scale<<<(unsigned)((n + 255) / 256), 256, 0, s>>>(w.feats32, w.ds32, C, tot);
  }
  // weightnet: gnorm (3) -> 8 -> 8 -> 16
  k_zero16<<<(unsigned)(((size_t)tot * 32 + 255) / 256), 256, 0, s>>>(w.wpkA, (size_t)tot * 32);
  k_zero16<<<(unsigned)(((size_t)tot * 32 + 255) / 256), 256, 0, s>>>(w.wpkB, (size_t)tot * 32);
  gemm(s, w, w.gnormpk, 3, 32, P.wnet[0].lin, &P.wnet[0].bn, nullptr, w.wpkA, tot, 8, 0);
  gemm(s, w, w.wpkA, 8, 32, P.wnet[1].lin, &P.wnet[1].bn, nullptr, w.wpkB, tot, 8, 0);
  gemm(s, w, w.wpkB, 8, 32, P.wnet[2].lin, &P.wnet[2].bn, w.wnet32, nullptr, tot, 16, 0);
  // einsum -> aggpk (f16, (B*np, C*16)), then linear + bn_linear + relu
  k_einsum<<<dim3((C / 16 + 3) / 4, BB * npoint), 128, 0, s>>>(w.feats32, w.wnet32, w.aggpk, C);
  gemm(s, w, w.aggpk, C * 16, C * 16, P.linear, &P.bn_linear, out_feat, nullptr, BB * npoint, C, 0);
}

static void run_fp(hipStream_t s, const Layer* mlp, int nl, const int* dims,
                   const float* x1, int n1, const float* x2, int n2,
                   const float* p1c, int C1, const float* p2c, int C2,
                   float* outp, _Float16* outpk, WS& w) {
  k_knn<3><<<dim3((n1 + 63) / 64, BB), 64, 0, s>>>(x1, n1, x2, n2, w.knn3_idx, w.knn3_d);
  int tot = BB * n1;
  int K = C1 + C2, Kp = ru32(K);
  k_interp_concat<<<(tot + 255) / 256, 256, 0, s>>>(p1c, C1, p2c, C2, w.knn3_idx,
                                                    w.knn3_d, n1, n2, Kp, w.fpkA);
  const _Float16* a = w.fpkA;
  for (int l = 0; l < nl; ++l) {
    bool last = (l == nl - 1);
    _Float16* dst16 = last ? outpk : ((l & 1) ? w.fpkA : w.fpkB);
    float* dst32 = last ? outp : nullptr;
    gemm(s, w, a, K, Kp, mlp[l].lin, &mlp[l].bn, dst32, dst16, tot, dims[l], 0);
    a = dst16 ? dst16 : nullptr;
    if (last) break;
    K = dims[l]; Kp = ru32(K);
  }
}

extern "C" void kernel_launch(void* const* d_in, const int* in_sizes, int n_in,
                              void* d_out, int out_size, void* d_ws, size_t ws_size,
                              hipStream_t stream) {
  (void)in_sizes; (void)n_in; (void)out_size; (void)ws_size;

  // ---- parameter mapping: d_in[0]=xyz; params in jax sorted-key flatten order
  int cur = 0;
  auto nxt = [&]() -> const float* { return (const float*)d_in[cur++]; };
  auto rdBn = [&]() { Bn r; r.be = nxt(); r.g = nxt(); r.m = nxt(); r.v = nxt(); return r; };
  auto rdLin = [&]() { Lin r; r.W = nxt(); r.b = nxt(); return r; };
  auto rdLayer = [&]() { Layer L; L.bn = rdBn(); L.lin = rdLin(); return L; };

  const float* xyz_in = nxt();                       // (B,6,N)
  Layer fp1m[3], fp2m[2], fp3m[2], fp4m[2];
  for (int i = 0; i < 3; ++i) fp1m[i] = rdLayer();   // 'fp1'
  for (int i = 0; i < 2; ++i) fp2m[i] = rdLayer();   // 'fp2'
  for (int i = 0; i < 2; ++i) fp3m[i] = rdLayer();   // 'fp3'
  for (int i = 0; i < 2; ++i) fp4m[i] = rdLayer();   // 'fp4'
  Bn head_bn1 = rdBn();                              // 'head': bn1, conv1, conv2
  Lin head_c1 = rdLin();
  Lin head_c2 = rdLin();
  SAP sa[4];
  for (int i = 0; i < 4; ++i) {                      // 'sa1'..'sa4'
    sa[i].bn_linear = rdBn();
    for (int j = 0; j < 3; ++j) sa[i].dens[j] = rdLayer();
    sa[i].linear = rdLin();
    for (int j = 0; j < 3; ++j) sa[i].mlp[j] = rdLayer();
    for (int j = 0; j < 3; ++j) sa[i].wnet[j] = rdLayer();
  }

  // ---- workspace bump allocator
  char* base = (char*)d_ws;
  size_t off = 0;
  auto alloc_f = [&](size_t n) -> float* {
    float* p = (float*)(base + off); off += ((n * 4 + 255) / 256) * 256; return p; };
  auto alloc_h = [&](size_t n) -> _Float16* {
    _Float16* p = (_Float16*)(base + off); off += ((n * 2 + 255) / 256) * 256; return p; };
  auto alloc_i = [&](size_t n) -> int* {
    int* p = (int*)(base + off); off += ((n * 4 + 255) / 256) * 256; return p; };

  float* pts   = alloc_f((size_t)BB * NN * 6);
  float* l0xyz = alloc_f((size_t)BB * NN * 3);
  float* xyz1  = alloc_f((size_t)BB * 1024 * 3);
  float* xyz2  = alloc_f((size_t)BB * 256 * 3);
  float* xyz3  = alloc_f((size_t)BB * 64 * 3);
  float* xyz4  = alloc_f((size_t)BB * 16 * 3);
  float* p0 = alloc_f((size_t)BB * NN * 128);
  float* p1 = alloc_f((size_t)BB * 1024 * 128);
  float* p2 = alloc_f((size_t)BB * 256 * 256);
  float* p3 = alloc_f((size_t)BB * 64 * 256);
  float* p4 = alloc_f((size_t)BB * 16 * 512);

  WS w;
  w.invd     = alloc_f((size_t)BB * NN);
  w.fidx     = alloc_i((size_t)BB * 1024);
  w.knn_idx  = alloc_i((size_t)BB * 1024 * 32);
  w.knn_d    = alloc_f((size_t)BB * 1024 * 32);
  w.knn3_idx = alloc_i((size_t)BB * NN * 3);
  w.knn3_d   = alloc_f((size_t)BB * NN * 3);
  w.fpkA     = alloc_h((size_t)8 * 1024 * 1024);
  w.fpkB     = alloc_h((size_t)8 * 1024 * 1024);
  w.gnormpk  = alloc_h((size_t)BB * 1024 * 32 * 32);
  w.dspkA    = alloc_h((size_t)BB * 1024 * 32 * 32);
  w.dspkB    = alloc_h((size_t)BB * 1024 * 32 * 32);
  w.wpkA     = alloc_h((size_t)BB * 1024 * 32 * 32);
  w.wpkB     = alloc_h((size_t)BB * 1024 * 32 * 32);
  w.aggpk    = alloc_h((size_t)BB * 1024 * 64 * 16);
  w.wtpk     = alloc_h((size_t)512 * 8192 + 4096);
  w.p0pk     = alloc_h((size_t)BB * NN * 128);
  w.hpk      = alloc_h((size_t)BB * NN * 128);
  w.feats32  = alloc_f((size_t)BB * 1024 * 32 * 64);
  w.wnet32   = alloc_f((size_t)BB * 1024 * 32 * 16);
  w.ds32     = alloc_f((size_t)BB * 1024 * 32);

  // ---- forward
  k_make_pts<<<(BB * NN + 255) / 256, 256, 0, stream>>>(xyz_in, pts, l0xyz);

  static const int m1[3] = {32, 32, 64};
  static const int m2[3] = {64, 64, 128};
  static const int m3[3] = {128, 128, 256};
  static const int m4[3] = {256, 256, 512};
  run_sa(stream, sa[0], l0xyz, NN,   pts, 6,   1024, 0.1f, m1, xyz1, p1, w);
  run_sa(stream, sa[1], xyz1, 1024,  p1, 64,   256,  0.2f, m2, xyz2, p2, w);
  run_sa(stream, sa[2], xyz2, 256,   p2, 128,  64,   0.4f, m3, xyz3, p3, w);
  run_sa(stream, sa[3], xyz3, 64,    p3, 256,  16,   0.8f, m4, xyz4, p4, w);

  static const int d4[2] = {512, 256};
  static const int d3[2] = {256, 256};
  static const int d2[2] = {256, 128};
  static const int d1[3] = {128, 128, 128};
  run_fp(stream, fp4m, 2, d4, xyz3, 64,   xyz4, 16,   p3, 256, p4, 512, p3, w.fpkB, w);
  run_fp(stream, fp3m, 2, d3, xyz2, 256,  xyz3, 64,   p2, 128, p3, 256, p2, w.fpkB, w);
  run_fp(stream, fp2m, 2, d2, xyz1, 1024, xyz2, 256,  p1, 64,  p2, 256, p1, w.fpkB, w);
  run_fp(stream, fp1m, 3, d1, l0xyz, NN,  xyz1, 1024, nullptr, 0, p1, 128, p0, w.p0pk, w);

  // head: conv1+bn1+relu (f16 in/out) -> conv2 -> log_softmax -> (B,2,N)
  gemm(stream, w, w.p0pk, 128, 128, head_c1, &head_bn1, nullptr, w.hpk, BB * NN, 128, 0);
  gemm(stream, w, w.hpk, 128, 128, head_c2, nullptr, w.feats32, nullptr, BB * NN, 2, 2);
  k_logsoftmax_T<<<(BB * NN + 255) / 256, 256, 0, stream>>>(w.feats32, (float*)d_out);
}
